// SRUEncoder_9955734192229
// MI455X (gfx1250) — compile-verified, matrix operand order checked by reading, not run on hardware
//
#include <hip/hip_runtime.h>
#include <hip/hip_bf16.h>

// ---------------------------------------------------------------------------
// SRU encoder: B=32, L=2048, D=512, NL=4
//   per layer:  U = h @ W[l]   (M=L*B=65536, K=512, N=1536)  -- bf16 WMMA,
//               A/B tiles staged to LDS with global_load_async_to_lds_b128
//               (double buffered, s_wait_asynccnt pipeline)
//               sequential scan over t (16384 independent (b,d) chains)
//   out = sum over layers of c_last  (B*D = 16384 f32)
// ---------------------------------------------------------------------------

#define BB 32
#define LL 2048
#define DD 512
#define NLAYERS 4
#define NN (3 * DD)   // 1536
#define LDSW 40       // LDS row stride in shorts (80 B = 5*16 B, conflict-free)

typedef __attribute__((ext_vector_type(16))) __bf16 bf16x16;
typedef __attribute__((ext_vector_type(8)))  float  v8f;

__device__ __forceinline__ unsigned short f2bf(float f) {
    unsigned int u = __float_as_uint(f);
    unsigned int r = (u + 0x7FFFu + ((u >> 16) & 1u)) >> 16; // RNE
    return (unsigned short)r;
}
__device__ __forceinline__ float bf2f(unsigned short s) {
    return __uint_as_float((unsigned int)s << 16);
}

// ---------------------------------------------------------------------------
// Transpose x (B,L,D) f32 -> h (L,B,D) bf16
// ---------------------------------------------------------------------------
__global__ __launch_bounds__(256) void transpose_x(const float* __restrict__ x,
                                                   unsigned short* __restrict__ h) {
    size_t idx = (size_t)blockIdx.x * 256 + threadIdx.x; // over L*B*D
    int d = (int)(idx & (DD - 1));
    size_t r = idx >> 9;            // t*B + b
    int b = (int)(r & (BB - 1));
    int t = (int)(r >> 5);
    h[idx] = f2bf(x[(((size_t)b * LL) + t) * DD + d]);
}

// ---------------------------------------------------------------------------
// W (NL, D, 3D) f32 -> W^T (NL, 3D, D) bf16  (so GEMM B tiles are row-major
// along K and can be async-copied to LDS without an in-kernel transpose)
// ---------------------------------------------------------------------------
__global__ __launch_bounds__(256) void conv_wT(const float* __restrict__ W,
                                               unsigned short* __restrict__ WT) {
    size_t idx = (size_t)blockIdx.x * 256 + threadIdx.x; // layout (l, n, k)
    int k = (int)(idx & (DD - 1));
    size_t r = idx >> 9;            // l*NN + n
    int n = (int)(r % NN);
    int l = (int)(r / NN);
    WT[idx] = f2bf(W[((size_t)l * DD + k) * NN + n]);
}

// ---------------------------------------------------------------------------
// Load a 16-element bf16 fragment from an LDS row, per the CDNA5 16-bit
// A/B matrix lane layout: lane half selects K-base 0 or 8; elements
// 0..7 = K kb..kb+7 ; 8..15 = K kb+16..kb+23 (row stride LDSW shorts).
// ---------------------------------------------------------------------------
__device__ __forceinline__ bf16x16 load_frag(const unsigned short* rowp, int kb) {
    union { unsigned int w[8]; bf16x16 v; } f;
    uint2 a = *(const uint2*)(rowp + kb);
    uint2 b = *(const uint2*)(rowp + kb + 4);
    uint2 c = *(const uint2*)(rowp + kb + 16);
    uint2 d = *(const uint2*)(rowp + kb + 20);
    f.w[0] = a.x; f.w[1] = a.y; f.w[2] = b.x; f.w[3] = b.y;
    f.w[4] = c.x; f.w[5] = c.y; f.w[6] = d.x; f.w[7] = d.y;
    return f.v;
}

// ---------------------------------------------------------------------------
// GEMM: C[M x 1536] = A[M x 512] (bf16) * Wt[1536 x 512]^T (bf16)
// block tile 128x128, 256 threads = 8 waves, wave tile 64x32 (4x2 WMMA tiles)
// Double-buffered LDS; tiles staged with per-lane async B128 loads to LDS.
// ---------------------------------------------------------------------------
__global__ __launch_bounds__(256) void gemm_u(const unsigned short* __restrict__ A,
                                              const unsigned short* __restrict__ Bt,
                                              float* __restrict__ C) {
    __shared__ unsigned short As[2][128 * LDSW];
    __shared__ unsigned short Bs[2][128 * LDSW];

    const int tid   = threadIdx.x;
    const int lane  = tid & 31;
    const int wave  = tid >> 5;
    const int rsel  = lane & 15;
    const int half  = lane >> 4;
    const int kb    = half * 8;

    const int blockM = blockIdx.y * 128;
    const int blockN = blockIdx.x * 128;
    const int waveM  = (wave >> 2) * 64; // 2 waves in M
    const int waveN  = (wave & 3) * 32;  // 4 waves in N

    const unsigned asBase = (unsigned)(uintptr_t)&As[0][0];
    const unsigned bsBase = (unsigned)(uintptr_t)&Bs[0][0];

    // Stage one 128x32 bf16 tile of A and of B^T into LDS buffer `buf` using
    // async B128 loads (16 B per lane per instruction, all 16 B aligned).
    auto stage = [&](int buf, int k0) {
#pragma unroll
        for (int i = 0; i < 2; ++i) {
            int q   = tid + i * 256;   // 0..511 8-short (16 B) chunks
            int row = q >> 2;          // 4 chunks per 32-k row
            int co  = (q & 3) * 8;     // short offset in row
            unsigned ldsOff = (unsigned)(buf * (128 * LDSW) + row * LDSW + co) * 2u;
            unsigned ldsA = asBase + ldsOff;
            unsigned offA = (unsigned)((blockM + row) * DD + k0 + co) * 2u;
            asm volatile("global_load_async_to_lds_b128 %0, %1, %2"
                         :: "v"(ldsA), "v"(offA), "s"(A) : "memory");
            unsigned ldsB = bsBase + ldsOff;
            unsigned offB = (unsigned)((blockN + row) * DD + k0 + co) * 2u;
            asm volatile("global_load_async_to_lds_b128 %0, %1, %2"
                         :: "v"(ldsB), "v"(offB), "s"(Bt) : "memory");
        }
    };

    stage(0, 0);
    asm volatile("s_wait_asynccnt 0x0" ::: "memory");
    __syncthreads();

    v8f acc[4][2] = {};
    for (int k0 = 0; k0 < DD; k0 += 32) {
        const int cur = (k0 >> 5) & 1;
        if (k0 + 32 < DD)
            stage(cur ^ 1, k0 + 32); // async DMA of next tiles overlaps WMMA

        const unsigned short* as = &As[cur][0];
        const unsigned short* bs = &Bs[cur][0];

        bf16x16 fa[4], fb[2];
#pragma unroll
        for (int i = 0; i < 4; ++i)
            fa[i] = load_frag(&as[(waveM + i * 16 + rsel) * LDSW], kb);
#pragma unroll
        for (int j = 0; j < 2; ++j)
            fb[j] = load_frag(&bs[(waveN + j * 16 + rsel) * LDSW], kb);

#pragma unroll
        for (int i = 0; i < 4; ++i)
#pragma unroll
            for (int j = 0; j < 2; ++j)
                acc[i][j] = __builtin_amdgcn_wmma_f32_16x16x32_bf16(
                    false, fa[i], false, fb[j], (short)0, acc[i][j], false, false);

        asm volatile("s_wait_asynccnt 0x0" ::: "memory");
        __syncthreads();
    }

    // ---- store: VGPR v -> M = v + 8*half within each 16x16 tile ----
    const int mbase = blockM + waveM + half * 8;
    const int nbase = blockN + waveN + rsel;
#pragma unroll
    for (int i = 0; i < 4; ++i)
#pragma unroll
        for (int j = 0; j < 2; ++j)
#pragma unroll
            for (int v = 0; v < 8; ++v)
                C[(size_t)(mbase + i * 16 + v) * NN + nbase + j * 16] = acc[i][j][v];
}

// ---------------------------------------------------------------------------
// SRU recurrence over one time chunk. One thread per (b,d) chain.
// U chunk layout: (Tc, B, 3D) f32. h layout: (L, B, D) bf16, updated in place.
// ---------------------------------------------------------------------------
__global__ __launch_bounds__(256) void sru_scan(const float* __restrict__ U,
                                                const int* __restrict__ mask,
                                                const float* __restrict__ state,
                                                const float* __restrict__ vc,
                                                const float* __restrict__ bias,
                                                unsigned short* __restrict__ hio,
                                                float* __restrict__ cbuf,
                                                float* __restrict__ csum,
                                                int layer, int t0, int Tc,
                                                int writeH, int accum) {
    const int idx = blockIdx.x * 256 + threadIdx.x; // 0..16383
    const int d = idx & (DD - 1);
    const int b = idx >> 9;

    const float vf = vc[layer * 2 * DD + d];
    const float vr = vc[layer * 2 * DD + DD + d];
    const float bf = bias[layer * 2 * DD + d];
    const float br = bias[layer * 2 * DD + DD + d];

    float c = (t0 == 0) ? state[layer * DD + d] : cbuf[idx];
    const int* mrow = &mask[b * LL + t0];

    for (int tt = 0; tt < Tc; tt += 8) {
        float u0[8], u1[8], u2[8], xv[8];
        int mk[8];
#pragma unroll
        for (int j = 0; j < 8; ++j) { // burst loads: 40 outstanding per thread
            size_t ub = ((size_t)(tt + j) * BB + b) * NN + d;
            u0[j] = U[ub];
            u1[j] = U[ub + DD];
            u2[j] = U[ub + 2 * DD];
            xv[j] = bf2f(hio[((size_t)(t0 + tt + j) * BB + b) * DD + d]);
            mk[j] = mrow[tt + j];
        }
#pragma unroll
        for (int j = 0; j < 8; ++j) {
            bool pad = (mk[j] == 0);
            float f = 1.f / (1.f + __expf(-(u1[j] + vf * c + bf)));
            float cn = f * c + (1.f - f) * u0[j];
            c = pad ? c : cn;
            float r = 1.f / (1.f + __expf(-(u2[j] + vr * c + br)));
            float ht = r * c + (1.f - r) * xv[j];
            ht = pad ? 0.f : ht;
            if (writeH)
                hio[((size_t)(t0 + tt + j) * BB + b) * DD + d] = f2bf(ht);
        }
    }
    cbuf[idx] = c;
    if (accum)
        csum[idx] += c;
}

// ---------------------------------------------------------------------------
extern "C" void kernel_launch(void* const* d_in, const int* in_sizes, int n_in,
                              void* d_out, int out_size, void* d_ws, size_t ws_size,
                              hipStream_t stream) {
    const float* x     = (const float*)d_in[0];
    const int*   mask  = (const int*)d_in[1];
    const float* state = (const float*)d_in[2];
    const float* W     = (const float*)d_in[3];
    const float* vc    = (const float*)d_in[4];
    const float* bias  = (const float*)d_in[5];

    // workspace carve-up (all offsets 256B-aligned)
    const size_t hBytes = (size_t)LL * BB * DD * sizeof(unsigned short);      // 67 MB
    const size_t wBytes = (size_t)NLAYERS * DD * NN * sizeof(unsigned short); // 6.3 MB
    const size_t cBytes = (size_t)BB * DD * sizeof(float);                    // 64 KB
    char* wsb = (char*)d_ws;
    unsigned short* hbuf = (unsigned short*)wsb;
    unsigned short* WbfT = (unsigned short*)(wsb + hBytes);
    float*          cbuf = (float*)(wsb + hBytes + wBytes);
    float*          Ubuf = (float*)(wsb + hBytes + wBytes + cBytes);

    // pick largest time-chunk that fits in the workspace
    const size_t fixed = hBytes + wBytes + cBytes;
    const int cands[] = {2048, 1024, 512, 256, 128, 64, 32, 16, 8};
    int Tc = 8;
    for (int i = 0; i < 9; ++i) {
        size_t need = fixed + (size_t)cands[i] * BB * NN * sizeof(float);
        if (need <= ws_size) { Tc = cands[i]; break; }
    }

    hipMemsetAsync(d_out, 0, (size_t)BB * DD * sizeof(float), stream);

    transpose_x<<<(LL * BB * DD) / 256, 256, 0, stream>>>(x, hbuf);
    conv_wT<<<(NLAYERS * DD * NN) / 256, 256, 0, stream>>>(W, WbfT);

    const int nch = LL / Tc;
    for (int l = 0; l < NLAYERS; ++l) {
        for (int ck = 0; ck < nch; ++ck) {
            int t0 = ck * Tc;
            const unsigned short* Achunk = hbuf + (size_t)t0 * BB * DD;
            const unsigned short* Wl = WbfT + (size_t)l * NN * DD;
            dim3 g(NN / 128, (Tc * BB) / 128);
            gemm_u<<<g, 256, 0, stream>>>(Achunk, Wl, Ubuf);
            sru_scan<<<(BB * DD) / 256, 256, 0, stream>>>(
                Ubuf, mask, state, vc, bias, hbuf, cbuf, (float*)d_out,
                l, t0, Tc, (l < NLAYERS - 1) ? 1 : 0, (ck == nch - 1) ? 1 : 0);
        }
    }
}
